// MixtureOfExperts_85401129713915
// MI455X (gfx1250) — compile-verified, matrix operand order checked by reading, not run on hardware
//
#include <hip/hip_runtime.h>

typedef _Float16 f16;
typedef __attribute__((ext_vector_type(16))) _Float16 v16h;
typedef __attribute__((ext_vector_type(8)))  float    v8f;
typedef __attribute__((ext_vector_type(4)))  _Float16 v4h;

#define B_TOK  4096
#define D_IN   1024
#define D_HID  4096
#define D_OUT  1024
#define NE     8
#define NSLOT  (B_TOK * 2)

#define BM 128
#define BN 128
#define BK 64
#define LDA 80   // BK + 16 halves pad -> 160B row stride (32B multiple)

// ---------------- CDNA5 async-copy to LDS via inline asm ----------------
// (builtin exists but its params use the unspellable CUDA __device__ LangAS;
//  inline asm sidesteps Sema's address-space checks entirely)
#define USE_ASYNC 1

typedef __attribute__((address_space(3))) f16 lds_f16;

#if __has_builtin(__builtin_amdgcn_s_wait_asynccnt)
#define WAIT_ASYNC(n) __builtin_amdgcn_s_wait_asynccnt(n)
#else
#define WAIT_ASYNC(n) asm volatile("s_wait_asynccnt " #n ::: "memory")
#endif

__device__ __forceinline__ void async_cp_b128(const f16* g, f16* l) {
    unsigned loff = (unsigned)(size_t)(lds_f16*)l;          // LDS byte offset
    unsigned long long ga = (unsigned long long)(size_t)g;  // global address
    asm volatile("global_load_async_to_lds_b128 %0, %1, off"
                 :: "v"(loff), "v"(ga) : "memory");
}

// ---------------------------------------------------------------- init / gate
__global__ void moe_init(int* meta) {
    if (threadIdx.x < 64) meta[threadIdx.x] = 0;
}

// meta (ints): [0..7]=counts, [16..23]=fill, [32..40]=base (exclusive scan)
__global__ void moe_gate(const float* __restrict__ x, const float* __restrict__ Wg,
                         const float* __restrict__ bg, const int* __restrict__ task_p,
                         int* __restrict__ meta, int* __restrict__ tok_e,
                         float* __restrict__ tok_w) {
    const int lane = threadIdx.x & 31;
    const int wave = threadIdx.x >> 5;
    const int t    = blockIdx.x * 8 + wave;          // one wave per token
    const int task = *task_p;
    const float* wg = Wg + (size_t)task * NE * D_IN;
    const float* xr = x + (size_t)t * D_IN;

    float acc[NE];
#pragma unroll
    for (int e = 0; e < NE; e++) acc[e] = 0.f;
    for (int j = lane; j < D_IN; j += 32) {
        float xv = xr[j];
#pragma unroll
        for (int e = 0; e < NE; e++) acc[e] += xv * wg[e * D_IN + j];
    }
#pragma unroll
    for (int e = 0; e < NE; e++) {
#pragma unroll
        for (int off = 16; off > 0; off >>= 1) acc[e] += __shfl_xor(acc[e], off, 32);
    }
    if (lane == 0) {
        float lg[NE];
#pragma unroll
        for (int e = 0; e < NE; e++) lg[e] = acc[e] + bg[task * NE + e];
        int i0 = 0;                                   // top-2, first-index ties
#pragma unroll
        for (int e = 1; e < NE; e++) if (lg[e] > lg[i0]) i0 = e;
        int i1 = (i0 == 0) ? 1 : 0;
#pragma unroll
        for (int e = 0; e < NE; e++)
            if (e != i0 && e != i1 && lg[e] > lg[i1]) i1 = e;
        float e1  = __expf(lg[i1] - lg[i0]);
        float inv = 1.f / (1.f + e1);
        tok_e[t * 2]     = i0;  tok_e[t * 2 + 1] = i1;
        tok_w[t * 2]     = inv; tok_w[t * 2 + 1] = e1 * inv;
        atomicAdd(&meta[i0], 1);
        atomicAdd(&meta[i1], 1);
    }
}

__global__ void moe_prefix(int* meta) {
    if (threadIdx.x == 0) {
        int run = 0;
        meta[32] = 0;
        for (int e = 0; e < NE; e++) { run += meta[e]; meta[33 + e] = run; }
    }
}

__global__ void moe_scatter(const int* __restrict__ tok_e, int* __restrict__ meta,
                            int* __restrict__ slot_token, int* __restrict__ tok_slot) {
    int t = blockIdx.x * blockDim.x + threadIdx.x;
#pragma unroll
    for (int k = 0; k < 2; k++) {
        int e   = tok_e[t * 2 + k];
        int pos = atomicAdd(&meta[16 + e], 1);
        int s   = meta[32 + e] + pos;
        slot_token[s]       = t;
        tok_slot[t * 2 + k] = s;
    }
}

// ----------------------------------------------------- f32 -> f16 pre-passes
__global__ void cvt_x(const float* __restrict__ x, f16* __restrict__ xh) {
    int i = blockIdx.x * blockDim.x + threadIdx.x;   // one float4 per thread
    float4 v = ((const float4*)x)[i];
    v4h o = {(f16)v.x, (f16)v.y, (f16)v.z, (f16)v.w};
    ((v4h*)xh)[i] = o;
}

// W: [E][K][N] f32  ->  Wt: [E][N][K] f16   (32x32 LDS tile transpose)
__global__ __launch_bounds__(256) void cvt_wT(const float* __restrict__ W,
                                              f16* __restrict__ Wt, int K, int N) {
    __shared__ f16 tile[32][33];
    const int e  = blockIdx.z;
    const int kb = blockIdx.y * 32, nb = blockIdx.x * 32;
    const float* Ws = W  + (size_t)e * K * N;
    f16*         Ts = Wt + (size_t)e * N * K;
    int r  = threadIdx.x >> 3;         // k row 0..31
    int c0 = (threadIdx.x & 7) * 4;    // 4 consecutive n
    float4 v = *(const float4*)&Ws[(size_t)(kb + r) * N + nb + c0];
    tile[c0 + 0][r] = (f16)v.x; tile[c0 + 1][r] = (f16)v.y;
    tile[c0 + 2][r] = (f16)v.z; tile[c0 + 3][r] = (f16)v.w;
    __syncthreads();
    int rn = threadIdx.x >> 3;         // n row 0..31
    int ck = (threadIdx.x & 7) * 4;    // 4 consecutive k
    v4h o = {tile[rn][ck], tile[rn][ck + 1], tile[rn][ck + 2], tile[rn][ck + 3]};
    *(v4h*)&Ts[(size_t)(nb + rn) * K + kb + ck] = o;
}

// --------------------------------------------------------- grouped WMMA GEMM
// out[slot, :] = relu( A[row(slot), :] @ Bt[e]^T + bias[e] )
// A: [*, KD] f16 (rows via rowmap or identity-in-slot-space); Bt: [NE][ND][KD] f16
template <int KD, int ND, bool OUT16>
__global__ __launch_bounds__(256) void moe_gemm(
    const f16* __restrict__ A, const f16* __restrict__ Bt,
    const float* __restrict__ bias, const int* __restrict__ meta,
    const int* __restrict__ rowmap, f16* __restrict__ out16,
    float* __restrict__ out32) {
    const int e    = blockIdx.z;
    const int rows = meta[e];
    const int m0   = blockIdx.x * BM;
    if (m0 >= rows) return;
    const int sb  = meta[32 + e];
    const int n0  = blockIdx.y * BN;
    const int tid = threadIdx.x;

    __shared__ f16 sA[2][BM][LDA];
    __shared__ f16 sB[2][BN][LDA];
    __shared__ int sRow[BM];

    if (tid < BM) {
        int m = m0 + tid;
        if (m >= rows) m = rows - 1;                 // clamp: padded rows never stored
        sRow[tid] = rowmap ? rowmap[sb + m] : (sb + m);
    }
    __syncthreads();

    // per-thread copy chunks: 4x16B for A, 4x16B for B per K-tile
    size_t aOff[4], bOff[4];
    f16 *aL[4], *bL[4];
#pragma unroll
    for (int i = 0; i < 4; i++) {
        int idx = i * 256 + tid;
        int r = idx >> 3, c = (idx & 7) * 8;
        aOff[i] = (size_t)sRow[r] * KD + c;
        aL[i]   = &sA[0][r][c];
        bOff[i] = ((size_t)e * ND + n0 + r) * KD + c;
        bL[i]   = &sB[0][r][c];
    }

    const int lane = tid & 31, wave = tid >> 5;
    const int wm = wave >> 1, wn = wave & 1;         // 4x2 waves: 32x64 per wave
    const int lh = lane >> 4, ll = lane & 15;

    v8f acc[2][4] = {};

    auto issue = [&](int k0, int bf) {
#pragma unroll
        for (int i = 0; i < 4; i++) {
#if USE_ASYNC
            async_cp_b128(A + aOff[i] + k0, aL[i] + bf * (BM * LDA));
            async_cp_b128(Bt + bOff[i] + k0, bL[i] + bf * (BN * LDA));
#else
            *(uint4*)(aL[i] + bf * (BM * LDA)) = *(const uint4*)(A + aOff[i] + k0);
            *(uint4*)(bL[i] + bf * (BN * LDA)) = *(const uint4*)(Bt + bOff[i] + k0);
#endif
        }
    };

    constexpr int NT = KD / BK;
    issue(0, 0);
    for (int t = 0; t < NT; ++t) {
        const int bf = t & 1;
        if (t + 1 < NT) {
            issue((t + 1) * BK, bf ^ 1);             // prefetch next tile
#if USE_ASYNC
            WAIT_ASYNC(8);                           // oldest 8 (this tile) done
#endif
        } else {
#if USE_ASYNC
            WAIT_ASYNC(0);
#endif
        }
        __syncthreads();
#pragma unroll
        for (int kk = 0; kk < BK / 32; kk++) {
            v16h a[2], b[4];
            const int akb = kk * 32 + lh * 16;
#pragma unroll
            for (int i = 0; i < 2; i++)
                a[i] = *(const v16h*)&sA[bf][wm * 32 + i * 16 + ll][akb];
#pragma unroll
            for (int j = 0; j < 4; j++)
                b[j] = *(const v16h*)&sB[bf][wn * 64 + j * 16 + ll][akb];
#pragma unroll
            for (int i = 0; i < 2; i++)
#pragma unroll
                for (int j = 0; j < 4; j++)
                    acc[i][j] = __builtin_amdgcn_wmma_f32_16x16x32_f16(
                        false, a[i], false, b[j], (short)0, acc[i][j], false, false);
        }
        __syncthreads();
    }

    // epilogue: bias + relu (C layout: vgpr v -> row v+8*lh, lane%16 -> col)
#pragma unroll
    for (int j = 0; j < 4; j++) {
        int n = n0 + wn * 64 + j * 16 + ll;
        float bv = bias[e * ND + n];
#pragma unroll
        for (int i = 0; i < 2; i++)
#pragma unroll
            for (int v = 0; v < 8; v++) {
                int m = m0 + wm * 32 + i * 16 + v + 8 * lh;
                if (m < rows) {
                    float r = fmaxf(acc[i][j][v] + bv, 0.f);
                    if constexpr (OUT16)
                        out16[(size_t)(sb + m) * ND + n] = (f16)r;
                    else
                        out32[(size_t)(sb + m) * ND + n] = r;
                }
            }
    }
}

// ------------------------------------------------------------------- combine
__global__ void moe_combine(const float* __restrict__ O, const int* __restrict__ tok_slot,
                            const float* __restrict__ tok_w, float* __restrict__ out) {
    int t = blockIdx.x;
    int c = threadIdx.x;                             // 256 x float4 = 1024 cols
    int s0 = tok_slot[t * 2], s1 = tok_slot[t * 2 + 1];
    float w0 = tok_w[t * 2], w1 = tok_w[t * 2 + 1];
    const float4* o0 = (const float4*)(O + (size_t)s0 * D_OUT);
    const float4* o1 = (const float4*)(O + (size_t)s1 * D_OUT);
    float4 a = o0[c], b = o1[c], r;
    r.x = w0 * a.x + w1 * b.x;
    r.y = w0 * a.y + w1 * b.y;
    r.z = w0 * a.z + w1 * b.z;
    r.w = w0 * a.w + w1 * b.w;
    ((float4*)(out + (size_t)t * D_OUT))[c] = r;
}

// -------------------------------------------------------------------- launch
extern "C" void kernel_launch(void* const* d_in, const int* in_sizes, int n_in,
                              void* d_out, int out_size, void* d_ws, size_t ws_size,
                              hipStream_t stream) {
    (void)in_sizes; (void)n_in; (void)out_size; (void)ws_size;
    const float* x   = (const float*)d_in[0];
    const float* Wg  = (const float*)d_in[1];
    const float* bg  = (const float*)d_in[2];
    const float* W1  = (const float*)d_in[3];
    const float* b1  = (const float*)d_in[4];
    const float* W2  = (const float*)d_in[5];
    const float* b2  = (const float*)d_in[6];
    const int*   tk  = (const int*)d_in[7];
    float* out = (float*)d_out;

    char* ws = (char*)d_ws;                          // needs ~240 MB
    int*   meta       = (int*)ws;                                    // 64 ints
    int*   slot_token = (int*)(ws + 256);
    int*   tok_slot   = (int*)(ws + 256 + 1 * NSLOT * 4);
    int*   tok_e      = (int*)(ws + 256 + 2 * NSLOT * 4);
    float* tok_w      = (float*)(ws + 256 + 3 * NSLOT * 4);
    size_t off = 1 << 18;
    f16*   xh  = (f16*)(ws + off);  off += (size_t)B_TOK * D_IN * 2;
    off = (off + 255) & ~(size_t)255;
    f16*   W1t = (f16*)(ws + off);  off += (size_t)NE * D_IN * D_HID * 2;
    f16*   W2t = (f16*)(ws + off);  off += (size_t)NE * D_HID * D_OUT * 2;
    f16*   H   = (f16*)(ws + off);  off += (size_t)NSLOT * D_HID * 2;
    float* O   = (float*)(ws + off);

    cvt_x  <<<(B_TOK * D_IN / 4) / 256, 256, 0, stream>>>(x, xh);
    cvt_wT <<<dim3(D_HID / 32, D_IN / 32, NE), 256, 0, stream>>>(W1, W1t, D_IN, D_HID);
    cvt_wT <<<dim3(D_OUT / 32, D_HID / 32, NE), 256, 0, stream>>>(W2, W2t, D_HID, D_OUT);
    moe_init   <<<1, 64, 0, stream>>>(meta);
    moe_gate   <<<B_TOK / 8, 256, 0, stream>>>(x, Wg, bg, tk, meta, tok_e, tok_w);
    moe_prefix <<<1, 32, 0, stream>>>(meta);
    moe_scatter<<<B_TOK / 256, 256, 0, stream>>>(tok_e, meta, slot_token, tok_slot);
    moe_gemm<D_IN, D_HID, true>
        <<<dim3(B_TOK / BM, D_HID / BN, NE), 256, 0, stream>>>(
            xh, W1t, b1, meta, slot_token, H, nullptr);
    moe_gemm<D_HID, D_OUT, false>
        <<<dim3(B_TOK / BM, D_OUT / BN, NE), 256, 0, stream>>>(
            H, W2t, b2, meta, nullptr, nullptr, O);
    moe_combine<<<B_TOK, 256, 0, stream>>>(O, tok_slot, tok_w, out);
}
